// WebAttackGNN_47390669144203
// MI455X (gfx1250) — matrix operand order repaired and checked by reading, not compiled, not run on hardware
//
#include <hip/hip_runtime.h>
#include <hip/hip_bf16.h>
#include <math.h>

typedef __attribute__((ext_vector_type(16))) __bf16 v16bf;
typedef __attribute__((ext_vector_type(8)))  float  v8f;

#define NEG_SLOPE 0.2f

// ---------------------------------------------------------------- utilities
__global__ __launch_bounds__(256) void fill_kernel(float* __restrict__ p, float v, long long n) {
  long long i = (long long)blockIdx.x * blockDim.x + threadIdx.x;
  if (i < n) p[i] = v;
}

__device__ __forceinline__ void atomicMaxF(float* addr, float val) {
  // ordered-int trick; valid for buffers initialized to -INF
  if (val >= 0.0f) atomicMax((int*)addr, __float_as_int(val));
  else             atomicMin((unsigned int*)addr, __float_as_uint(val));
}

__device__ __forceinline__ unsigned int pack_bf16x2(float a, float b) {
  __bf16 x = (__bf16)a, y = (__bf16)b;
  unsigned short ux = __builtin_bit_cast(unsigned short, x);
  unsigned short uy = __builtin_bit_cast(unsigned short, y);
  return ((unsigned int)uy << 16) | (unsigned int)ux;
}

// ---------------------------------------------------------------- WMMA GEMM
// C[M,N] = A[M,K] @ B[K,N]; fp32 in/out, bf16 WMMA compute (f32 accumulate).
// Block tile 64(M) x 32(N), 8 waves (4x2), K stepped by 64 (2 WMMA / iter).
// LDS holds tiles already in WMMA fragment order: each lane's 16 bf16
// fragment is 32 contiguous bytes -> wide ds_load_b128 pairs.
__global__ __launch_bounds__(256) void gemm_bf16_wmma_kernel(
    const float* __restrict__ A, const float* __restrict__ B,
    float* __restrict__ C, int M, int K, int N) {
  // A: 4 rowTiles x 2 kTiles, B: 2 colTiles x 2 kTiles; 256 u32 (1KB) per tile
  __shared__ unsigned int AsU[8 * 256];
  __shared__ unsigned int BsU[4 * 256];
  const int tid  = threadIdx.x;
  const int wave = tid >> 5;
  const int lane = tid & 31;
  const int rowBase = blockIdx.x * 64;
  const int colBase = blockIdx.y * 32;
  const int rowTile = wave >> 1;            // 0..3
  const int colTile = wave & 1;             // 0..1
  v8f acc = {};

  for (int k0 = 0; k0 < K; k0 += 64) {
    // ---- stage A (64 rows x 64 k) as bf16 pairs in fragment order ----
    for (int i = tid; i < 2048; i += 256) {
      int r  = i >> 5;                      // 0..63
      int kp = (i & 31) << 1;               // even k within 0..62
      int kT = kp >> 5, kk = kp & 31;
      int m = r & 15, rT = r >> 4;
      int fl = m | ((kk & 8) << 1);                       // fragment lane
      int j  = (kk & 7) | ((kk & 16) >> 1);               // fragment elem (even)
      int gr = rowBase + r;
      float f0 = 0.f, f1 = 0.f;
      if (gr < M) {
        const float2 av = *(const float2*)(A + (size_t)gr * K + (k0 + kp));
        f0 = av.x; f1 = av.y;
      }
      AsU[(rT * 2 + kT) * 256 + fl * 8 + (j >> 1)] = pack_bf16x2(f0, f1);
    }
    // ---- stage B (64 k x 32 cols) ----
    for (int i = tid; i < 1024; i += 256) {
      int kp = (i >> 5) << 1;               // even k
      int c  = i & 31;
      int kT = kp >> 5, kk = kp & 31;
      int cT = c >> 4, cc = c & 15;
      int fl = cc | ((kk & 8) << 1);                      // fragment lane
      int j  = (kk & 7) | ((kk & 16) >> 1);               // fragment elem (even)
      float f0 = B[(size_t)(k0 + kp) * N + (colBase + c)];
      float f1 = B[(size_t)(k0 + kp + 1) * N + (colBase + c)];
      BsU[(cT * 2 + kT) * 256 + fl * 8 + (j >> 1)] = pack_bf16x2(f0, f1);
    }
    __syncthreads();

    const v16bf a0 = *(const v16bf*)&AsU[(rowTile * 2 + 0) * 256 + lane * 8];
    const v16bf a1 = *(const v16bf*)&AsU[(rowTile * 2 + 1) * 256 + lane * 8];
    const v16bf b0 = *(const v16bf*)&BsU[(colTile * 2 + 0) * 256 + lane * 8];
    const v16bf b1 = *(const v16bf*)&BsU[(colTile * 2 + 1) * 256 + lane * 8];
    acc = __builtin_amdgcn_wmma_f32_16x16x32_bf16(false, a0, false, b0,
                                                  (short)0, acc, false, false);
    acc = __builtin_amdgcn_wmma_f32_16x16x32_bf16(false, a1, false, b1,
                                                  (short)0, acc, false, false);
    __syncthreads();
  }

  // C/D layout: VGPR r -> M = r (lanes 0-15) or r+8 (lanes 16-31), N = lane&15
  const int col  = colBase + (colTile << 4) + (lane & 15);
  const int rAdd = (lane < 16) ? 0 : 8;
  const int row0 = rowBase + (rowTile << 4) + rAdd;
  float* cp = C + (size_t)row0 * N + col;
  if (rowBase + 64 <= M) {      // block-uniform fast path: no per-lane guards
#pragma unroll
    for (int r = 0; r < 8; ++r) cp[(size_t)r * N] = acc[r];
  } else {
#pragma unroll
    for (int r = 0; r < 8; ++r)
      if (row0 + r < M) cp[(size_t)r * N] = acc[r];
  }
}

// ---------------------------------------------------------------- attention
// s_src[n,h] = sum_c h[n,h,c]*a_src[h,c]; s_dst likewise. One wave per (n,h).
__global__ __launch_bounds__(256) void attn_logits_kernel(
    const float* __restrict__ h, const float* __restrict__ a_src,
    const float* __restrict__ a_dst, float* __restrict__ sS,
    float* __restrict__ sD, long long N, int H, int C) {
  long long w = (long long)blockIdx.x * 8 + (threadIdx.x >> 5);
  int lane = threadIdx.x & 31;
  if (w >= N * H) return;
  long long n = w / H;
  int hh = (int)(w % H);
  const float* hrow = h + ((size_t)n * H + hh) * C;
  const float* as = a_src + (size_t)hh * C;
  const float* ad = a_dst + (size_t)hh * C;
  float ps = 0.f, pd = 0.f;
  for (int c = lane; c < C; c += 32) {
    float v = hrow[c];
    ps += v * as[c];
    pd += v * ad[c];
  }
#pragma unroll
  for (int off = 16; off > 0; off >>= 1) {
    ps += __shfl_xor(ps, off, 32);
    pd += __shfl_xor(pd, off, 32);
  }
  if (lane == 0) { sS[w] = ps; sD[w] = pd; }
}

// e = leaky_relu(sS[src]+sD[dst]); segment max into mm
__global__ __launch_bounds__(256) void edge_logits_kernel(
    const long long* __restrict__ src, const long long* __restrict__ dst,
    const float* __restrict__ sS, const float* __restrict__ sD,
    float* __restrict__ ee, float* __restrict__ mm, long long E, int H) {
  long long t = (long long)blockIdx.x * blockDim.x + threadIdx.x;
  if (t >= E * H) return;
  long long e = t / H;
  int hh = (int)(t % H);
  long long s = src[e], d = dst[e];
  float v = sS[s * H + hh] + sD[d * H + hh];
  v = (v > 0.f) ? v : NEG_SLOPE * v;
  ee[t] = v;
  atomicMaxF(&mm[d * H + hh], v);
}

// ee := exp(e - m[dst]); segment sum into dd
__global__ __launch_bounds__(256) void edge_exp_kernel(
    const long long* __restrict__ dst, float* __restrict__ ee,
    const float* __restrict__ mm, float* __restrict__ dd, long long E, int H) {
  long long t = (long long)blockIdx.x * blockDim.x + threadIdx.x;
  if (t >= E * H) return;
  long long e = t / H;
  int hh = (int)(t % H);
  long long d = dst[e];
  float v = expf(ee[t] - mm[d * H + hh]);
  ee[t] = v;
  atomicAdd(&dd[d * H + hh], v);
}

// out[dst,h,:] += h[src,h,:] * (ex / (denom[dst,h]+1e-16)); float4 per thread
__global__ __launch_bounds__(256) void aggregate_kernel(
    const long long* __restrict__ src, const long long* __restrict__ dst,
    const float* __restrict__ h, const float* __restrict__ ee,
    const float* __restrict__ dd, float* __restrict__ out,
    long long E, int H, int C) {
  const int q = C >> 2;
  long long t = (long long)blockIdx.x * blockDim.x + threadIdx.x;
  long long total = E * (long long)H * q;
  if (t >= total) return;
  int c4 = (int)(t % q) * 4;
  long long eh = t / q;
  long long e = eh / H;
  int hh = (int)(eh % H);
  long long s = src[e], d = dst[e];
  float alpha = ee[eh] / (dd[d * H + hh] + 1e-16f);
  const float4 hv = *(const float4*)(h + ((size_t)s * H + hh) * C + c4);
  float* op = out + ((size_t)d * H + hh) * C + c4;
  atomicAdd(op + 0, hv.x * alpha);
  atomicAdd(op + 1, hv.y * alpha);
  atomicAdd(op + 2, hv.z * alpha);
  atomicAdd(op + 3, hv.w * alpha);
}

// out = elu(in + bias) elementwise, bias broadcast over columns
__global__ __launch_bounds__(256) void bias_elu_kernel(
    const float* __restrict__ in, const float* __restrict__ bias,
    float* __restrict__ out, long long total, int Ccols) {
  long long i = (long long)blockIdx.x * blockDim.x + threadIdx.x;
  if (i >= total) return;
  int c = (int)(i % Ccols);
  float v = in[i] + bias[c];
  out[i] = (v > 0.f) ? v : expm1f(v);
}

// logits = emb @ Wf2 + bf2; log_softmax over 6 classes
__global__ __launch_bounds__(256) void head_kernel(
    const float* __restrict__ emb, const float* __restrict__ Wf2,
    const float* __restrict__ bf2, float* __restrict__ out, long long N) {
  long long n = (long long)blockIdx.x * blockDim.x + threadIdx.x;
  if (n >= N) return;
  float lg[6];
#pragma unroll
  for (int j = 0; j < 6; ++j) lg[j] = bf2[j];
  const float* er = emb + (size_t)n * 64;
  for (int k = 0; k < 64; ++k) {
    float v = er[k];
#pragma unroll
    for (int j = 0; j < 6; ++j) lg[j] += v * Wf2[k * 6 + j];
  }
  float mx = lg[0];
#pragma unroll
  for (int j = 1; j < 6; ++j) mx = fmaxf(mx, lg[j]);
  float s = 0.f;
#pragma unroll
  for (int j = 0; j < 6; ++j) s += expf(lg[j] - mx);
  float lse = mx + logf(s);
#pragma unroll
  for (int j = 0; j < 6; ++j) out[n * 6 + j] = lg[j] - lse;
}

// ---------------------------------------------------------------- launcher
static inline unsigned int gdiv(long long a, int b) {
  return (unsigned int)((a + b - 1) / b);
}

extern "C" void kernel_launch(void* const* d_in, const int* in_sizes, int n_in,
                              void* d_out, int out_size, void* d_ws, size_t ws_size,
                              hipStream_t stream) {
  const float*     x   = (const float*)d_in[0];
  const long long* ei  = (const long long*)d_in[1];   // int64 [2,E]
  const float*     W1  = (const float*)d_in[2];
  const float*     a1s = (const float*)d_in[3];
  const float*     a1d = (const float*)d_in[4];
  const float*     b1  = (const float*)d_in[5];
  const float*     W2  = (const float*)d_in[6];
  const float*     a2s = (const float*)d_in[7];
  const float*     a2d = (const float*)d_in[8];
  const float*     b2  = (const float*)d_in[9];
  const float*     Wf1 = (const float*)d_in[10];
  const float*     bf1 = (const float*)d_in[11];
  const float*     Wf2 = (const float*)d_in[12];
  const float*     bf2 = (const float*)d_in[13];

  const int D = 768, H1 = 4, C = 128, HC = 512, F = 64, NCL = 6;
  const long long N = in_sizes[0] / D;
  const long long E = in_sizes[1] / 2;
  const long long* src = ei;
  const long long* dst = ei + E;

  // workspace layout (floats)
  float* bufA = (float*)d_ws;                 // h1 -> h2 -> ff_raw
  float* bufB = bufA + (size_t)N * HC;        // out1 -> out2
  float* sS   = bufB + (size_t)N * HC;
  float* sD   = sS + (size_t)N * H1;
  float* mm   = sD + (size_t)N * H1;
  float* dd   = mm + (size_t)N * H1;
  float* ee   = dd + (size_t)N * H1;          // E*H1 floats

  float* out_logits = (float*)d_out;
  float* out_emb    = out_logits + (size_t)N * NCL;

  // ===================== GAT layer 1 (H=4, C=128) =====================
  fill_kernel<<<gdiv(N * HC, 256), 256, 0, stream>>>(bufB, 0.f, N * HC);
  fill_kernel<<<gdiv(N * H1, 256), 256, 0, stream>>>(mm, -INFINITY, N * H1);
  fill_kernel<<<gdiv(N * H1, 256), 256, 0, stream>>>(dd, 0.f, N * H1);

  dim3 g1(gdiv(N, 64), HC / 32);
  gemm_bf16_wmma_kernel<<<g1, 256, 0, stream>>>(x, W1, bufA, (int)N, D, HC);

  attn_logits_kernel<<<gdiv(N * H1, 8), 256, 0, stream>>>(bufA, a1s, a1d, sS, sD, N, H1, C);
  edge_logits_kernel<<<gdiv(E * H1, 256), 256, 0, stream>>>(src, dst, sS, sD, ee, mm, E, H1);
  edge_exp_kernel<<<gdiv(E * H1, 256), 256, 0, stream>>>(dst, ee, mm, dd, E, H1);
  aggregate_kernel<<<gdiv(E * H1 * (C / 4), 256), 256, 0, stream>>>(src, dst, bufA, ee, dd, bufB, E, H1, C);
  bias_elu_kernel<<<gdiv(N * HC, 256), 256, 0, stream>>>(bufB, b1, bufB, N * HC, HC);

  // ===================== GAT layer 2 (H=1, C=128) =====================
  dim3 g2(gdiv(N, 64), C / 32);
  gemm_bf16_wmma_kernel<<<g2, 256, 0, stream>>>(bufB, W2, bufA, (int)N, HC, C);

  fill_kernel<<<gdiv(N * C, 256), 256, 0, stream>>>(bufB, 0.f, N * C);
  fill_kernel<<<gdiv(N, 256), 256, 0, stream>>>(mm, -INFINITY, N);
  fill_kernel<<<gdiv(N, 256), 256, 0, stream>>>(dd, 0.f, N);

  attn_logits_kernel<<<gdiv(N, 8), 256, 0, stream>>>(bufA, a2s, a2d, sS, sD, N, 1, C);
  edge_logits_kernel<<<gdiv(E, 256), 256, 0, stream>>>(src, dst, sS, sD, ee, mm, E, 1);
  edge_exp_kernel<<<gdiv(E, 256), 256, 0, stream>>>(dst, ee, mm, dd, E, 1);
  aggregate_kernel<<<gdiv(E * (C / 4), 256), 256, 0, stream>>>(src, dst, bufA, ee, dd, bufB, E, 1, C);
  bias_elu_kernel<<<gdiv(N * C, 256), 256, 0, stream>>>(bufB, b2, bufB, N * C, C);

  // ===================== FF head =====================
  dim3 g3(gdiv(N, 64), F / 32);
  gemm_bf16_wmma_kernel<<<g3, 256, 0, stream>>>(bufB, Wf1, bufA, (int)N, C, F);
  bias_elu_kernel<<<gdiv(N * F, 256), 256, 0, stream>>>(bufA, bf1, out_emb, N * F, F);
  head_kernel<<<gdiv(N, 256), 256, 0, stream>>>(out_emb, Wf2, bf2, out_logits, N);
}
